// AutoCorrelation_10660108828964
// MI455X (gfx1250) — matrix operand order repaired and checked by reading, not compiled, not run on hardware
//
#include <hip/hip_runtime.h>
#include <math.h>

// ---------------------------------------------------------------------------
// AutoCorrelation (Autoformer) for MI455X / gfx1250.
// Correlation-mean reformulated as block-diagonal bf16 WMMA GEMM.
// B=16, L=2048, H=8, E=64 -> C=H*E=512, top_k = int(log(2048)) = 7.
// ---------------------------------------------------------------------------

typedef __attribute__((ext_vector_type(16))) __bf16       v16bf;
typedef __attribute__((ext_vector_type(8)))  float        v8f;
typedef __attribute__((ext_vector_type(4)))  float        f32x4;
typedef __attribute__((ext_vector_type(4)))  __bf16       bf16x4;
typedef __attribute__((ext_vector_type(4)))  unsigned int u32x4;

constexpr int BB   = 16;
constexpr int LL   = 2048;
constexpr int CC   = 8 * 64;      // H*E = 512 (contiguous innermost dims of [B,L,H,E])
constexpr int KTOP = 7;           // int(1 * log(2048))
constexpr int LBLK = LL / 16;     // 128 row/col blocks of 16

union FragBF {
    v16bf  v;
    u32x4  q[2];    // two 16-byte chunks of 8 bf16
    bf16x4 b4[4];   // four 4-element bf16 chunks
};

// ---------------------------------------------------------------------------
// Zero the mean-accumulator.
// ---------------------------------------------------------------------------
__global__ void zero_kernel(float* __restrict__ p, int n) {
    int i = blockIdx.x * blockDim.x + threadIdx.x;
    if (i < n) p[i] = 0.0f;
}

// ---------------------------------------------------------------------------
// fp32 -> bf16 conversion (4 elements / thread).
// ---------------------------------------------------------------------------
__global__ void cvt_bf16_kernel(const float* __restrict__ src, __bf16* __restrict__ dst, int n4) {
    int gid = blockIdx.x * blockDim.x + threadIdx.x;
    if (gid < n4) {
        f32x4 x = *(const f32x4*)(src + 4 * (size_t)gid);
        *(bf16x4*)(dst + 4 * (size_t)gid) = __builtin_convertvector(x, bf16x4);
    }
}

// ---------------------------------------------------------------------------
// Block-diagonal Gram accumulation via v_wmma_f32_16x16x32_bf16.
//   workgroup = (D, b), D = block diagonal index in [0,128).
//   All tiles (I, J=(I-D) mod 128) map to the same 31 tau values
//   tau = 16*D + (i - j), i,j in [0,16), so every wave chains its WMMAs
//   into ONE f32 accumulator, then a 31-slot LDS diagonal reduction.
// ---------------------------------------------------------------------------
template <bool PRECONV>
__global__ __launch_bounds__(256)
void corr_diag_kernel(const float*  __restrict__ qf, const float*  __restrict__ kf,
                      const __bf16* __restrict__ qb, const __bf16* __restrict__ kb,
                      float* __restrict__ mean_sum) {
    __shared__ float ldsDiag[31];

    const int tid  = threadIdx.x;
    const int wave = tid >> 5;          // 8 waves (wave32)
    const int lane = tid & 31;
    const int m    = lane & 15;         // A row (M) / B,C column (N)
    const int hi   = lane >> 4;         // lane-half select (ISA fragment layout)
    const int D    = blockIdx.x;        // block diagonal
    const int b    = blockIdx.y;        // batch

    if (tid < 31) ldsDiag[tid] = 0.0f;
    __syncthreads();

    v8f acc = {};                        // 16x16 f32 accumulator (8 VGPRs)

    for (int ii = 0; ii < LBLK / 8; ++ii) {
        const int I = wave + 8 * ii;                   // row block
        const int J = (I - D) & (LBLK - 1);            // col block on diagonal D
        const size_t qrow = ((size_t)b * LL + 16 * I + m) * CC;
        const size_t krow = ((size_t)b * LL + 16 * J + m) * CC;

#pragma unroll 4
        for (int kc = 0; kc < CC; kc += 32) {
            FragBF A, Bv;
            if (PRECONV) {
                const __bf16* qa = qb + qrow + kc;
                const __bf16* ka = kb + krow + kc;
                // A (16x32 bf16): lane half hi holds K in {8hi..8hi+7} U {16+8hi..23+8hi}
                A.q[0]  = *(const u32x4*)(qa + 8 * hi);
                A.q[1]  = *(const u32x4*)(qa + 16 + 8 * hi);
                // B (32x16 bf16) = K^T: lane half hi holds K in {8hi..8hi+7} U {16+8hi..23+8hi}
                Bv.q[0] = *(const u32x4*)(ka + 8 * hi);
                Bv.q[1] = *(const u32x4*)(ka + 16 + 8 * hi);
            } else {
                const float* qa = qf + qrow + kc;
                const float* ka = kf + krow + kc;
                A.b4[0]  = __builtin_convertvector(*(const f32x4*)(qa + 8 * hi),       bf16x4);
                A.b4[1]  = __builtin_convertvector(*(const f32x4*)(qa + 8 * hi + 4),   bf16x4);
                A.b4[2]  = __builtin_convertvector(*(const f32x4*)(qa + 16 + 8 * hi),  bf16x4);
                A.b4[3]  = __builtin_convertvector(*(const f32x4*)(qa + 20 + 8 * hi),  bf16x4);
                Bv.b4[0] = __builtin_convertvector(*(const f32x4*)(ka + 8 * hi),       bf16x4);
                Bv.b4[1] = __builtin_convertvector(*(const f32x4*)(ka + 8 * hi + 4),   bf16x4);
                Bv.b4[2] = __builtin_convertvector(*(const f32x4*)(ka + 16 + 8 * hi),  bf16x4);
                Bv.b4[3] = __builtin_convertvector(*(const f32x4*)(ka + 20 + 8 * hi),  bf16x4);
            }
            // D = A x B + C  (v_wmma_f32_16x16x32_bf16)
            acc = __builtin_amdgcn_wmma_f32_16x16x32_bf16(
                false, A.v, false, Bv.v, (short)0, acc, false, false);
        }
    }

    // C/D layout: element r at lane -> (i = r + 8*hi, j = lane&15); tau_local = i-j+15.
#pragma unroll
    for (int r = 0; r < 8; ++r) {
        const int i = r + 8 * hi;
        atomicAdd(&ldsDiag[i - m + 15], acc[r]);
    }
    __syncthreads();

    if (tid < 31) {
        const int tau = (16 * D + tid - 15) & (LL - 1);
        atomicAdd(&mean_sum[(size_t)b * LL + tau], ldsDiag[tid]);
    }
}

// ---------------------------------------------------------------------------
// Per-batch top-7 over mean_sum[b, :], then softmax of the (scaled) scores.
// ---------------------------------------------------------------------------
__global__ __launch_bounds__(256)
void topk_kernel(const float* __restrict__ mean_sum,
                 float* __restrict__ wsW, int* __restrict__ wsD) {
    const int b   = blockIdx.x;
    const int tid = threadIdx.x;
    __shared__ float sv[256];
    __shared__ int   si[256];
    __shared__ int   chI[KTOP];
    __shared__ float chV[KTOP];

    for (int it = 0; it < KTOP; ++it) {
        float best = -3.4e38f;
        int   bi   = 0;
        for (int t = tid; t < LL; t += 256) {
            float val = mean_sum[(size_t)b * LL + t];
            bool skip = false;
            for (int j = 0; j < it; ++j) skip |= (chI[j] == t);
            if (!skip && val > best) { best = val; bi = t; }
        }
        sv[tid] = best; si[tid] = bi;
        __syncthreads();
        for (int s = 128; s > 0; s >>= 1) {
            if (tid < s && sv[tid + s] > sv[tid]) { sv[tid] = sv[tid + s]; si[tid] = si[tid + s]; }
            __syncthreads();
        }
        if (tid == 0) { chI[it] = si[0]; chV[it] = sv[0] * (1.0f / (float)CC); }  // mean over H*E
        __syncthreads();
    }

    if (tid == 0) {
        float mx = chV[0];
        for (int j = 1; j < KTOP; ++j) mx = fmaxf(mx, chV[j]);
        float e[KTOP], s = 0.0f;
        for (int j = 0; j < KTOP; ++j) { e[j] = __expf(chV[j] - mx); s += e[j]; }
        const float inv = 1.0f / s;
        for (int j = 0; j < KTOP; ++j) { wsW[b * 8 + j] = e[j] * inv; wsD[b * 8 + j] = chI[j]; }
    }
}

// ---------------------------------------------------------------------------
// out[b,t,:,:] = sum_k w[b,k] * V[b,(t+d[b,k]) mod L,:,:]   (float4 streaming)
// ---------------------------------------------------------------------------
__global__ __launch_bounds__(256)
void agg_kernel(const float* __restrict__ v, const float* __restrict__ wsW,
                const int* __restrict__ wsD, float* __restrict__ out) {
    const size_t gid = (size_t)blockIdx.x * 256 + threadIdx.x;   // over B*L*(C/4)
    const int    c4  = (int)(gid % (CC / 4));
    const size_t bt  = gid / (CC / 4);
    const int    t   = (int)(bt % LL);
    const int    b   = (int)(bt / LL);

    float w[KTOP]; int d[KTOP];
#pragma unroll
    for (int j = 0; j < KTOP; ++j) { w[j] = wsW[b * 8 + j]; d[j] = wsD[b * 8 + j]; }

    f32x4 acc = {0.f, 0.f, 0.f, 0.f};
#pragma unroll
    for (int j = 0; j < KTOP; ++j) {
        const int tt = (t + d[j]) & (LL - 1);
        const f32x4 x = *(const f32x4*)(v + (((size_t)b * LL + tt) * CC + 4 * c4));
        acc += w[j] * x;
    }
    *(f32x4*)(out + 4 * gid) = acc;
}

// ---------------------------------------------------------------------------
extern "C" void kernel_launch(void* const* d_in, const int* in_sizes, int n_in,
                              void* d_out, int out_size, void* d_ws, size_t ws_size,
                              hipStream_t stream) {
    const float* q = (const float*)d_in[0];
    const float* k = (const float*)d_in[1];
    const float* v = (const float*)d_in[2];
    // d_in[3] = attn_mask (unused, as in reference)
    float* out = (float*)d_out;

    const size_t N = (size_t)BB * LL * CC;       // 16,777,216 elements per tensor

    // Workspace layout
    float*  mean_sum = (float*)d_ws;             // B*L floats
    float*  wsW      = mean_sum + (size_t)BB * LL;
    int*    wsD      = (int*)(wsW + BB * 8);
    __bf16* qb       = (__bf16*)(wsD + BB * 8);  // 16B-aligned (offset 132096 bytes)
    __bf16* kb       = qb + N;
    const size_t baseBytes = (size_t)BB * LL * 4 + (size_t)BB * 8 * 4 + (size_t)BB * 8 * 4;
    const bool preconv = ws_size >= baseBytes + 2 * N * sizeof(__bf16);

    zero_kernel<<<(BB * LL + 255) / 256, 256, 0, stream>>>(mean_sum, BB * LL);

    const dim3 grid(LBLK, BB);   // (128 diagonals, 16 batches)
    if (preconv) {
        const int n4 = (int)(N / 4);
        cvt_bf16_kernel<<<(n4 + 255) / 256, 256, 0, stream>>>(q, qb, n4);
        cvt_bf16_kernel<<<(n4 + 255) / 256, 256, 0, stream>>>(k, kb, n4);
        corr_diag_kernel<true><<<grid, 256, 0, stream>>>(q, k, qb, kb, mean_sum);
    } else {
        corr_diag_kernel<false><<<grid, 256, 0, stream>>>(q, k, qb, kb, mean_sum);
    }

    topk_kernel<<<BB, 256, 0, stream>>>(mean_sum, wsW, wsD);

    const size_t nOut4 = N / 4;                  // 4,194,304 float4s
    agg_kernel<<<(unsigned)(nOut4 / 256), 256, 0, stream>>>(v, wsW, wsD, out);
}
